// FCOS_54314156425693
// MI455X (gfx1250) — matrix operand order repaired and checked by reading, not compiled, not run on hardware
//
#include <hip/hip_runtime.h>

typedef __bf16 bf16_t;
typedef __attribute__((ext_vector_type(16))) __bf16 v16bf;
typedef __attribute__((ext_vector_type(8)))  float  v8f;
typedef __attribute__((ext_vector_type(4)))  unsigned int u32x4;
typedef __attribute__((ext_vector_type(8)))  int i32x8;
typedef __attribute__((ext_vector_type(4)))  int i32x4;

union FragU { uint4 u[2]; v16bf v; };
union PackU { __bf16 h[8]; uint4 u; };

#define TOTPOS   20267
#define REG_BASE 3242720   // 2*80*20267
#define CTR_BASE 3404856   // REG_BASE + 2*4*20267

// ---------------------------------------------------------------------------
// TDM: stage a (3 rows) x (66 cols) x (256 ch) bf16 tile from a zero-padded
// NHWC tensor into LDS with one tensor_load_to_lds.
//   dim0 = channel (256, contiguous), dim1 = x (stride 256), dim2 = y (stride
//   256*Wp).  LDS pad: after every 512B (128 dwords) insert 16B (4 dwords)
//   -> column stride 528B == s_in[...][264] layout.
//   tensor_dim1 = remaining cols from tile start => hardware zero-fills the
//   right-edge overhang (left/top/bottom are covered by the padded buffer).
// ---------------------------------------------------------------------------
__device__ __forceinline__ void tdm_stage_tile(const bf16_t* __restrict__ gsrc,
                                               int rem_cols, int row_stride_elems,
                                               void* lds_dst) {
  const unsigned long long ga = (unsigned long long)(uintptr_t)gsrc;
  const unsigned lds = (unsigned)(uintptr_t)lds_dst;   // low 32 bits = LDS offset
  u32x4 g0;
  g0[0] = 1u;                                              // count=1, user D#
  g0[1] = lds;                                             // lds_addr
  g0[2] = (unsigned)(ga & 0xFFFFFFFFull);                  // global_addr[31:0]
  g0[3] = (unsigned)((ga >> 32) & 0x1FFFFFFull) | (2u << 30); // addr[56:32] | type=2
  const unsigned s1 = (unsigned)row_stride_elems;          // y-stride (data_size units)
  i32x8 g1;
  g1[0] = (1 << 16) | (1 << 20) | (6 << 22) | (3 << 25);   // 2B elems, pad_en, 128dw/4dw
  g1[1] = (256 << 16);                                     // tensor_dim0 = 256
  g1[2] = (rem_cols & 0xFFFF) << 16;                       // tensor_dim1 = remaining x
  g1[3] = (256 << 16);                                     // tile_dim0 = 256
  g1[4] = 66 | (3 << 16);                                  // tile_dim1=66, tile_dim2=3
  g1[5] = 256;                                             // tensor_dim0_stride
  g1[6] = (int)((s1 & 0xFFFFu) << 16);                     // tensor_dim1_stride lo16
  g1[7] = (int)(s1 >> 16);                                 // tensor_dim1_stride hi
  i32x4 g2; g2[0] = 3; g2[1] = 0; g2[2] = 0; g2[3] = 0;    // tensor_dim2 = 3
  i32x4 g3; g3[0] = 0; g3[1] = 0; g3[2] = 0; g3[3] = 0;
#if __has_include(<hip/amd_detail/amd_gfx1250_TDM.h>)
  i32x8 g4; g4[0]=0; g4[1]=0; g4[2]=0; g4[3]=0; g4[4]=0; g4[5]=0; g4[6]=0; g4[7]=0;
  __builtin_amdgcn_tensor_load_to_lds(g0, g1, g2, g3, g4, 0);
#else
  __builtin_amdgcn_tensor_load_to_lds(g0, g1, g2, g3, 0);
#endif
}

// B fragment (32x16, K=ci, N=pos): lane<16 holds K 0..15 of col, lane>=16 K 16..31.
__device__ __forceinline__ v16bf load_bfrag(const bf16_t* __restrict__ s,
                                            int dy, int dx, int pt, int c0, int lane) {
  const int lm  = lane & 15;
  const int col = pt * 16 + lm + dx;           // 0..65
  const int ko  = c0 + ((lane < 16) ? 0 : 16);
  const bf16_t* p = s + (size_t)(dy * 66 + col) * 264 + ko;
  FragU f;
  f.u[0] = *(const uint4*)(p);
  f.u[1] = *(const uint4*)(p + 8);
  return f.v;
}

// A fragment from pre-packed weights: wf[((step*ntiles)+m)*512 + lane*16 + e]
__device__ __forceinline__ v16bf load_afrag(const bf16_t* __restrict__ wf,
                                            int step, int m, int ntiles, int lane) {
  const bf16_t* p = wf + ((size_t)(step * ntiles + m) << 9) + lane * 16;
  FragU f;
  f.u[0] = *(const uint4*)(p);
  f.u[1] = *(const uint4*)(p + 8);
  return f.v;
}

// ---------------------------------------------------------------------------
// Weight pre-pack kernels: OIHW f32 -> WMMA-A-fragment-ready bf16.
// A 16x32 k order used here (same as B frags): lane<16 holds K{0..15}, lane>=16 K{16..31}.
// ---------------------------------------------------------------------------
__device__ __forceinline__ int afrag_K(int lane, int e) {
  return (lane < 16) ? e : e + 16;
}

__global__ void pack_tower_w(const float* __restrict__ w, bf16_t* __restrict__ wf, int nlayers) {
  const int idx = blockIdx.x * blockDim.x + threadIdx.x;
  const int total = nlayers * 72 * 16 * 512;
  if (idx >= total) return;
  const int e = idx & 15, lane = (idx >> 4) & 31, m = (idx >> 9) & 15;
  const int step = (idx >> 13) % 72;
  const int l = idx / (72 * 16 * 512);
  const int c = step & 7, t = step >> 3;
  const int K = afrag_K(lane, e);
  const int co = m * 16 + (lane & 15), ci = c * 32 + K, ky = t / 3, kx = t % 3;
  wf[idx] = (bf16_t)w[((((size_t)l * 256 + co) * 256 + ci) * 3 + ky) * 3 + kx];
}

__global__ void pack_score_w(const float* __restrict__ w, bf16_t* __restrict__ wf) {
  const int idx = blockIdx.x * blockDim.x + threadIdx.x;
  const int total = 72 * 5 * 512;
  if (idx >= total) return;
  const int e = idx & 15, lane = (idx >> 4) & 31, m = (idx >> 9) % 5;
  const int step = idx / 2560;
  const int c = step & 7, t = step >> 3;
  const int K = afrag_K(lane, e);
  const int co = m * 16 + (lane & 15), ci = c * 32 + K, ky = t / 3, kx = t % 3;
  wf[idx] = (bf16_t)w[(((size_t)co * 256 + ci) * 3 + ky) * 3 + kx];
}

// pred(4ch) + ctr(1ch) fused into one zero-padded 16-channel tile.
__global__ void pack_pc_w(const float* __restrict__ wp, const float* __restrict__ wc,
                          bf16_t* __restrict__ wf) {
  const int idx = blockIdx.x * blockDim.x + threadIdx.x;
  const int total = 72 * 512;
  if (idx >= total) return;
  const int e = idx & 15, lane = (idx >> 4) & 31;
  const int step = idx >> 9;
  const int c = step & 7, t = step >> 3;
  const int K = afrag_K(lane, e);
  const int co = lane & 15, ci = c * 32 + K, ky = t / 3, kx = t % 3;
  float v = 0.f;
  if (co < 4)       v = wp[(((size_t)co * 256 + ci) * 3 + ky) * 3 + kx];
  else if (co == 4) v = wc[((size_t)ci * 3 + ky) * 3 + kx];
  wf[idx] = (bf16_t)v;
}

// f32 NCHW feature -> bf16 NHWC with 1-pixel zero halo: [(H+2)][(W+2)][256]
__global__ void nchw_to_nhwc_pad(const float* __restrict__ in, bf16_t* __restrict__ out,
                                 int H, int W) {
  const int Hp = H + 2, Wp = W + 2;
  const int idx = blockIdx.x * blockDim.x + threadIdx.x;
  const int total = 2 * 256 * Hp * Wp;
  if (idx >= total) return;
  const int c  = idx & 255;
  const int xp = (idx >> 8) % Wp;
  const int yp = (idx >> 8) / Wp % Hp;
  const int n  = idx / (256 * Wp * Hp);
  float v = 0.f;
  if (xp > 0 && xp <= W && yp > 0 && yp <= H)
    v = in[(((size_t)n * 256 + c) * H + (yp - 1)) * W + (xp - 1)];
  out[idx] = (bf16_t)v;
}

// Zero the halo ring of the three padded intermediate buffers (per level).
__global__ void zero_borders(bf16_t* __restrict__ a, bf16_t* __restrict__ b,
                             bf16_t* __restrict__ c, int H, int W) {
  const int Hp = H + 2, Wp = W + 2;
  const int idx = blockIdx.x * blockDim.x + threadIdx.x;
  const int total = 2 * Hp * Wp * 32;          // one uint4 (8 ch) per thread
  if (idx >= total) return;
  const int g   = idx & 31;
  const int pix = idx >> 5;
  const int xp  = pix % Wp;
  const int yp  = (pix / Wp) % Hp;
  if (xp != 0 && xp != Wp - 1 && yp != 0 && yp != Hp - 1) return;
  const size_t off = ((size_t)pix << 8) + (size_t)g * 8;
  const uint4 z = make_uint4(0u, 0u, 0u, 0u);
  *(uint4*)(a + off) = z;
  *(uint4*)(b + off) = z;
  *(uint4*)(c + off) = z;
}

// ---------------------------------------------------------------------------
// Tower conv layer: 3x3 SAME, 256->256, bias + ReLU. Padded bf16 NHWC in/out.
// Block = 256 threads (8 waves), one row of 64 output positions.
// Wave w computes co-tiles {2w, 2w+1} over 4 position sub-tiles.
// ---------------------------------------------------------------------------
__global__ void tower_conv_kernel(const bf16_t* __restrict__ in,
                                  const bf16_t* __restrict__ wf,
                                  const float* __restrict__ bias,
                                  bf16_t* __restrict__ out,
                                  int H, int W) {
  __shared__ __align__(16) bf16_t s_in[3 * 66 * 264];
  const int tid = threadIdx.x, lane = tid & 31, wave = tid >> 5;
  const int Hp = H + 2, Wp = W + 2;
  const int TW = (W + 63) >> 6;
  const int b = blockIdx.x;
  const int xbase = (b % TW) << 6;
  const int y = (b / TW) % H;
  const int n = b / (TW * H);

  if (wave == 0) {
    // padded coords: row y-1 -> y, col xbase-1 -> xbase
    tdm_stage_tile(in + ((size_t)(n * Hp + y) * Wp + xbase) * 256,
                   Wp - xbase, 256 * Wp, (void*)s_in);
    __builtin_amdgcn_s_wait_tensorcnt(0);
  }
  __syncthreads();

  v8f acc[2][4] = {};
  for (int step = 0; step < 72; ++step) {
    const int t = step >> 3, c0 = (step & 7) << 5;
    const int dy = t / 3, dx = t % 3;
    const v16bf a0 = load_afrag(wf, step, 2 * wave,     16, lane);
    const v16bf a1 = load_afrag(wf, step, 2 * wave + 1, 16, lane);
#pragma unroll
    for (int pt = 0; pt < 4; ++pt) {
      const v16bf bf = load_bfrag(s_in, dy, dx, pt, c0, lane);
      acc[0][pt] = __builtin_amdgcn_wmma_f32_16x16x32_bf16(false, a0, false, bf,
                                                           (short)0, acc[0][pt], false, false);
      acc[1][pt] = __builtin_amdgcn_wmma_f32_16x16x32_bf16(false, a1, false, bf,
                                                           (short)0, acc[1][pt], false, false);
    }
  }

  const int hi = (lane < 16) ? 0 : 8;
#pragma unroll
  for (int mi = 0; mi < 2; ++mi) {
    const int cobase = (2 * wave + mi) << 4;
    const float4 b0 = *(const float4*)(bias + cobase + hi);
    const float4 b1 = *(const float4*)(bias + cobase + hi + 4);
    const float bb[8] = {b0.x, b0.y, b0.z, b0.w, b1.x, b1.y, b1.z, b1.w};
#pragma unroll
    for (int pt = 0; pt < 4; ++pt) {
      const int x = xbase + pt * 16 + (lane & 15);
      if (x < W) {
        PackU pk;
#pragma unroll
        for (int r = 0; r < 8; ++r) {
          float v = acc[mi][pt][r] + bb[r];
          v = v > 0.f ? v : 0.f;
          pk.h[r] = (bf16_t)v;
        }
        *(uint4*)(out + (((size_t)(n * Hp + y + 1) * Wp + x + 1) << 8) + cobase + hi) = pk.u;
      }
    }
  }
}

// ---------------------------------------------------------------------------
// Score head: 256 -> 80 (5 co-tiles), bias, no ReLU, f32 scatter to d_out.
// Block = 160 threads (5 waves), wave m owns co-tile m.
// ---------------------------------------------------------------------------
__global__ void head_score_kernel(const bf16_t* __restrict__ in,
                                  const bf16_t* __restrict__ wf,
                                  const float* __restrict__ bias,
                                  float* __restrict__ out,
                                  int H, int W, int loff) {
  __shared__ __align__(16) bf16_t s_in[3 * 66 * 264];
  const int tid = threadIdx.x, lane = tid & 31, wave = tid >> 5;
  const int Hp = H + 2, Wp = W + 2;
  const int TW = (W + 63) >> 6;
  const int b = blockIdx.x;
  const int xbase = (b % TW) << 6;
  const int y = (b / TW) % H;
  const int n = b / (TW * H);

  if (wave == 0) {
    tdm_stage_tile(in + ((size_t)(n * Hp + y) * Wp + xbase) * 256,
                   Wp - xbase, 256 * Wp, (void*)s_in);
    __builtin_amdgcn_s_wait_tensorcnt(0);
  }
  __syncthreads();

  v8f acc[4] = {};
  for (int step = 0; step < 72; ++step) {
    const int t = step >> 3, c0 = (step & 7) << 5;
    const int dy = t / 3, dx = t % 3;
    const v16bf a = load_afrag(wf, step, wave, 5, lane);
#pragma unroll
    for (int pt = 0; pt < 4; ++pt) {
      const v16bf bf = load_bfrag(s_in, dy, dx, pt, c0, lane);
      acc[pt] = __builtin_amdgcn_wmma_f32_16x16x32_bf16(false, a, false, bf,
                                                        (short)0, acc[pt], false, false);
    }
  }

  const int hi = (lane < 16) ? 0 : 8;
  const int cobase = wave << 4;
#pragma unroll
  for (int pt = 0; pt < 4; ++pt) {
    const int x = xbase + pt * 16 + (lane & 15);
    if (x < W) {
      const int p = loff + y * W + x;
#pragma unroll
      for (int r = 0; r < 8; ++r) {
        const int co = cobase + hi + r;
        out[(size_t)(n * 80 + co) * TOTPOS + p] = acc[pt][r] + bias[co];
      }
    }
  }
}

// ---------------------------------------------------------------------------
// Pred(4, ReLU) + Ctr(1) head on one padded 16-channel co-tile.
// Block = 64 threads; wave 0 computes, wave 1 idles after barrier.
// ---------------------------------------------------------------------------
__global__ void head_pc_kernel(const bf16_t* __restrict__ in,
                               const bf16_t* __restrict__ wf,
                               const float* __restrict__ pred_b,
                               const float* __restrict__ ctr_b,
                               float* __restrict__ out,
                               int H, int W, int loff) {
  __shared__ __align__(16) bf16_t s_in[3 * 66 * 264];
  const int tid = threadIdx.x, lane = tid & 31, wave = tid >> 5;
  const int Hp = H + 2, Wp = W + 2;
  const int TW = (W + 63) >> 6;
  const int b = blockIdx.x;
  const int xbase = (b % TW) << 6;
  const int y = (b / TW) % H;
  const int n = b / (TW * H);

  if (wave == 0) {
    tdm_stage_tile(in + ((size_t)(n * Hp + y) * Wp + xbase) * 256,
                   Wp - xbase, 256 * Wp, (void*)s_in);
    __builtin_amdgcn_s_wait_tensorcnt(0);
  }
  __syncthreads();

  if (wave == 0) {
    v8f acc[4] = {};
    for (int step = 0; step < 72; ++step) {
      const int t = step >> 3, c0 = (step & 7) << 5;
      const int dy = t / 3, dx = t % 3;
      const v16bf a = load_afrag(wf, step, 0, 1, lane);
#pragma unroll
      for (int pt = 0; pt < 4; ++pt) {
        const v16bf bf = load_bfrag(s_in, dy, dx, pt, c0, lane);
        acc[pt] = __builtin_amdgcn_wmma_f32_16x16x32_bf16(false, a, false, bf,
                                                          (short)0, acc[pt], false, false);
      }
    }
    const int hi = (lane < 16) ? 0 : 8;
#pragma unroll
    for (int pt = 0; pt < 4; ++pt) {
      const int x = xbase + pt * 16 + (lane & 15);
      if (x < W) {
        const int p = loff + y * W + x;
#pragma unroll
        for (int r = 0; r < 8; ++r) {
          const int co = hi + r;
          if (co < 4) {
            float v = acc[pt][r] + pred_b[co];
            v = v > 0.f ? v : 0.f;
            out[REG_BASE + (size_t)(n * 4 + co) * TOTPOS + p] = v;
          } else if (co == 4) {
            out[CTR_BASE + (size_t)n * TOTPOS + p] = acc[pt][r] + ctr_b[0];
          }
        }
      }
    }
  }
}

// ---------------------------------------------------------------------------
extern "C" void kernel_launch(void* const* d_in, const int* in_sizes, int n_in,
                              void* d_out, int out_size, void* d_ws, size_t ws_size,
                              hipStream_t stream) {
  (void)in_sizes; (void)n_in; (void)out_size; (void)ws_size;
  const float* feat[5];
  for (int i = 0; i < 5; ++i) feat[i] = (const float*)d_in[i];
  const float* cls_w   = (const float*)d_in[5];
  const float* cls_b   = (const float*)d_in[6];
  const float* box_w   = (const float*)d_in[7];
  const float* box_b   = (const float*)d_in[8];
  const float* score_w = (const float*)d_in[9];
  const float* score_b = (const float*)d_in[10];
  const float* pred_w  = (const float*)d_in[11];
  const float* pred_b  = (const float*)d_in[12];
  const float* ctr_w   = (const float*)d_in[13];
  const float* ctr_b   = (const float*)d_in[14];
  float* out = (float*)d_out;

  const size_t TOWER_W_ELEMS = (size_t)4 * 72 * 16 * 512;    // 2,359,296
  const size_t SCORE_W_ELEMS = (size_t)72 * 5 * 512;         //   184,320
  const size_t PC_W_ELEMS    = (size_t)72 * 512;             //    36,864
  const size_t ACT_ELEMS     = (size_t)2 * 256 * 102 * 154;  // 8,042,496 (largest, padded)

  auto au = [](size_t v) { return (v + 255) & ~(size_t)255; };
  char* p = (char*)d_ws;
  bf16_t* wf_cls   = (bf16_t*)p; p += au(TOWER_W_ELEMS * 2);
  bf16_t* wf_box   = (bf16_t*)p; p += au(TOWER_W_ELEMS * 2);
  bf16_t* wf_score = (bf16_t*)p; p += au(SCORE_W_ELEMS * 2);
  bf16_t* wf_pc    = (bf16_t*)p; p += au(PC_W_ELEMS * 2);
  bf16_t* X  = (bf16_t*)p; p += au(ACT_ELEMS * 2);
  bf16_t* T0 = (bf16_t*)p; p += au(ACT_ELEMS * 2);
  bf16_t* T1 = (bf16_t*)p; p += au(ACT_ELEMS * 2);
  bf16_t* T2 = (bf16_t*)p; p += au(ACT_ELEMS * 2);

  pack_tower_w<<<(int)((TOWER_W_ELEMS + 255) / 256), 256, 0, stream>>>(cls_w, wf_cls, 4);
  pack_tower_w<<<(int)((TOWER_W_ELEMS + 255) / 256), 256, 0, stream>>>(box_w, wf_box, 4);
  pack_score_w<<<(int)((SCORE_W_ELEMS + 255) / 256), 256, 0, stream>>>(score_w, wf_score);
  pack_pc_w<<<(int)((PC_W_ELEMS + 255) / 256), 256, 0, stream>>>(pred_w, ctr_w, wf_pc);

  static const int HS[5]   = {100, 50, 25, 13, 7};
  static const int WSH[5]  = {152, 76, 38, 19, 10};
  static const int LOFF[5] = {0, 15200, 19000, 19950, 20197};

  for (int i = 0; i < 5; ++i) {
    const int H = HS[i], W = WSH[i];
    const int Hp = H + 2, Wp = W + 2;

    const int ctot = 2 * 256 * Hp * Wp;
    nchw_to_nhwc_pad<<<(ctot + 255) / 256, 256, 0, stream>>>(feat[i], X, H, W);
    const int ztot = 2 * Hp * Wp * 32;
    zero_borders<<<(ztot + 255) / 256, 256, 0, stream>>>(T0, T1, T2, H, W);

    const int nb = 2 * H * ((W + 63) / 64);
    const size_t LS = (size_t)72 * 16 * 512;   // per-layer packed weight stride

    // cls tower: X -> T0 -> T1 -> T0 -> T1
    tower_conv_kernel<<<nb, 256, 0, stream>>>(X,  wf_cls + 0 * LS, cls_b + 0,   T0, H, W);
    tower_conv_kernel<<<nb, 256, 0, stream>>>(T0, wf_cls + 1 * LS, cls_b + 256, T1, H, W);
    tower_conv_kernel<<<nb, 256, 0, stream>>>(T1, wf_cls + 2 * LS, cls_b + 512, T0, H, W);
    tower_conv_kernel<<<nb, 256, 0, stream>>>(T0, wf_cls + 3 * LS, cls_b + 768, T1, H, W);
    head_score_kernel<<<nb, 160, 0, stream>>>(T1, wf_score, score_b, out, H, W, LOFF[i]);

    // box tower: X -> T0 -> T2 -> T0 -> T2
    tower_conv_kernel<<<nb, 256, 0, stream>>>(X,  wf_box + 0 * LS, box_b + 0,   T0, H, W);
    tower_conv_kernel<<<nb, 256, 0, stream>>>(T0, wf_box + 1 * LS, box_b + 256, T2, H, W);
    tower_conv_kernel<<<nb, 256, 0, stream>>>(T2, wf_box + 2 * LS, box_b + 512, T0, H, W);
    tower_conv_kernel<<<nb, 256, 0, stream>>>(T0, wf_box + 3 * LS, box_b + 768, T2, H, W);
    head_pc_kernel<<<nb, 64, 0, stream>>>(T2, wf_pc, pred_b, ctr_b, out, H, W, LOFF[i]);
  }
}
